// CMRG_3126736191996
// MI455X (gfx1250) — compile-verified, matrix operand order checked by reading, not run on hardware
//
#include <hip/hip_runtime.h>
#include <hip/hip_bf16.h>
#include <math.h>

typedef __attribute__((ext_vector_type(16))) _Float16 v16h;
typedef __attribute__((ext_vector_type(8)))  _Float16 v8h;
typedef __attribute__((ext_vector_type(8)))  float    v8f;

namespace {
constexpr int cN   = 50000;
constexpr int cR   = 500;
constexpr int cM   = 120000;
constexpr int cE   = 960000;
constexpr int cNE  = 800000;
constexpr int cB   = 8192;
constexpr int cD   = 200;
constexpr int cH1  = 256;
constexpr int cHID = 512;
constexpr int cOC  = 50;
constexpr int kpad(int K) { return (K + 31) & ~31; }
}

// ---------------------------------------------------------------------------
// WMMA GEMM: C[M,N] = act( A[M,K] @ B[K,N] + bias + addend[perm] )
// Ah: f16 row-major, lda = Kp (K padded to 32, zero-filled)
// BT: f16 column-major [N,Kp] (BT[n*Kp+k] = B[k,n], zero-padded)
// One wave computes a 64x16 C macro-tile (4 stacked 16x16 WMMA tiles).
// No conditionals in the K-loop: all loads are aligned b128, OOB clamped.
// ---------------------------------------------------------------------------
__global__ void k_gemm_wmma(const _Float16* __restrict__ Ah, int lda,
                            const _Float16* __restrict__ BT,
                            float* __restrict__ C, int ldc,
                            int M, int N, int Kp,
                            const float* __restrict__ bias,
                            const float* __restrict__ addend,
                            const int* __restrict__ add_perm,
                            int relu)
{
  const int lane = threadIdx.x & 31;
  const int wave = threadIdx.x >> 5;
  const int tn  = blockIdx.x * 16;
  const int tm0 = (blockIdx.y * 4 + wave) * 64;

  const int half = lane >> 4;         // K-half select per ISA A/B layout
  const int l16  = lane & 15;
  const int bcol = tn + l16;
  const int bcol_c = (bcol < N) ? bcol : (N - 1);      // clamp for safe loads

  const _Float16* Bp = BT + (size_t)bcol_c * Kp;

  const _Float16* Ap[4];
#pragma unroll
  for (int s = 0; s < 4; ++s) {
    int r = tm0 + 16 * s;
    if (r + 16 > M) r = M - 16;                        // clamp (M is mult of 16)
    Ap[s] = Ah + (size_t)(r + l16) * lda;
  }

  v8f acc[4] = {};
  for (int k0 = 0; k0 < Kp; k0 += 32) {
    const int kb = k0 + half * 8;
    const v8h b0 = *(const v8h*)(Bp + kb);
    const v8h b1 = *(const v8h*)(Bp + kb + 16);
    const v16h bv = __builtin_shufflevector(b0, b1,
        0,1,2,3,4,5,6,7,8,9,10,11,12,13,14,15);
#pragma unroll
    for (int s = 0; s < 4; ++s) {
      const v8h a0 = *(const v8h*)(Ap[s] + kb);
      const v8h a1 = *(const v8h*)(Ap[s] + kb + 16);
      const v16h av = __builtin_shufflevector(a0, a1,
          0,1,2,3,4,5,6,7,8,9,10,11,12,13,14,15);
      acc[s] = __builtin_amdgcn_wmma_f32_16x16x32_f16(false, av, false, bv,
                                                      (short)0, acc[s],
                                                      false, false);
    }
  }

  if (bcol < N) {
    const float bia = bias ? bias[bcol] : 0.0f;
#pragma unroll
    for (int s = 0; s < 4; ++s) {
      const int rbase = tm0 + 16 * s;
      if (rbase < M) {
#pragma unroll
        for (int r = 0; r < 8; ++r) {
          const int row = rbase + half * 8 + r;  // C layout: VGPR r -> M=r+8*half
          float v = acc[s][r] + bia;
          if (addend) {
            const int ar = add_perm ? add_perm[row] : row;
            v += addend[(size_t)ar * ldc + bcol];
          }
          if (relu) v = fmaxf(v, 0.0f);
          C[(size_t)row * ldc + bcol] = v;
        }
      }
    }
  }
}

// f32 [rows,K] row-major -> f16 [rows,Kp] row-major, zero-padded
__global__ void k_cvt_f16_pad(const float* __restrict__ src,
                              _Float16* __restrict__ dst,
                              int rows, int K, int Kp)
{
  const int idx = blockIdx.x * blockDim.x + threadIdx.x;
  if (idx >= rows * Kp) return;
  const int r = idx / Kp, k = idx % Kp;
  dst[idx] = (k < K) ? (_Float16)src[(size_t)r * K + k] : (_Float16)0.0f;
}

// f32 B[K,N] row-major -> f16 BT[N,Kp] (column-major, zero-padded)
__global__ void k_prep_bT(const float* __restrict__ B, _Float16* __restrict__ BT,
                          int K, int N, int Kp)
{
  const int idx = blockIdx.x * blockDim.x + threadIdx.x;
  if (idx >= N * Kp) return;
  const int n = idx / Kp, k = idx % Kp;
  BT[idx] = (k < K) ? (_Float16)B[(size_t)k * N + n] : (_Float16)0.0f;
}

// ---------------------------------------------------------------------------
// Elementwise / scatter kernels (HBM-bound)
// ---------------------------------------------------------------------------
__global__ void k_l2norm(const float* __restrict__ x, float* __restrict__ y, int D)
{
  __shared__ float red[256];
  const int row = blockIdx.x;
  const int t = threadIdx.x;
  float xi = 0.0f, v = 0.0f;
  if (t < D) { xi = x[(size_t)row * D + t]; v = xi * xi; }
  red[t] = v; __syncthreads();
  for (int s = 128; s > 0; s >>= 1) { if (t < s) red[t] += red[t + s]; __syncthreads(); }
  const float nrm = fmaxf(sqrtf(red[0]), 1e-12f);
  if (t < D) y[(size_t)row * D + t] = xi / nrm;
}

__global__ void k_agg_x(const float* __restrict__ ent, const int* __restrict__ perm,
                        const float* __restrict__ rel,
                        const int* __restrict__ b_x, const int* __restrict__ b_gi,
                        const int* __restrict__ esrc, const int* __restrict__ edst,
                        const float* __restrict__ ew,
                        float* __restrict__ agg, int E)
{
  const int idx = blockIdx.x * blockDim.x + threadIdx.x;
  if (idx >= E * 400) return;
  const int e = idx / 400, c = idx % 400;
  const int s = esrc[e];
  float val;
  if (c < 200) {
    int n = b_x[s]; if (perm) n = perm[n];
    val = ent[(size_t)n * 200 + c];
  } else {
    val = rel[(size_t)b_gi[s] * 200 + (c - 200)];
  }
  atomicAdd(&agg[(size_t)edst[e] * 400 + c], val * ew[e]);
}

__global__ void k_agg(const float* __restrict__ h, int Cw,
                      const int* __restrict__ esrc, const int* __restrict__ edst,
                      const float* __restrict__ ew,
                      float* __restrict__ out, int E)
{
  const int idx = blockIdx.x * blockDim.x + threadIdx.x;
  if (idx >= E * Cw) return;
  const int e = idx / Cw, c = idx % Cw;
  atomicAdd(&out[(size_t)edst[e] * Cw + c], h[(size_t)esrc[e] * Cw + c] * ew[e]);
}

__global__ void k_count(const int* __restrict__ seg, float* __restrict__ cnt, int Mn)
{
  const int i = blockIdx.x * blockDim.x + threadIdx.x;
  if (i < Mn) atomicAdd(&cnt[seg[i]], 1.0f);
}

__global__ void k_scatter_add(const float* __restrict__ src, const int* __restrict__ seg,
                              float* __restrict__ dst, int Mn, int Cw)
{
  const int idx = blockIdx.x * blockDim.x + threadIdx.x;
  if (idx >= Mn * Cw) return;
  const int i = idx / Cw, c = idx % Cw;
  atomicAdd(&dst[(size_t)seg[i] * Cw + c], src[(size_t)i * Cw + c]);
}

__global__ void k_divcnt(float* __restrict__ mean, const float* __restrict__ cnt,
                         int Nr, int Cw)
{
  const int idx = blockIdx.x * blockDim.x + threadIdx.x;
  if (idx >= Nr * Cw) return;
  mean[idx] /= fmaxf(cnt[idx / Cw], 1.0f);
}

__global__ void k_score(const float* __restrict__ e, const float* __restrict__ mean,
                        const int* __restrict__ b_x, const float* __restrict__ Wle,
                        const float* __restrict__ ble, float* __restrict__ score, int Mn)
{
  const int i = blockIdx.x * blockDim.x + threadIdx.x;
  if (i >= Mn) return;
  const int n = b_x[i];
  const float* er = e + (size_t)n * 200;
  const float* mr = mean + (size_t)n * 200;
  float s = ble[0];
  for (int c = 0; c < 200; ++c) s += er[c] * Wle[c] + mr[c] * Wle[200 + c];
  score[i] = s;
}

__device__ __forceinline__ unsigned fenc(float x) {
  unsigned u = __float_as_uint(x);
  return (u & 0x80000000u) ? ~u : (u | 0x80000000u);
}
__device__ __forceinline__ float fdec(unsigned u) {
  return (u & 0x80000000u) ? __uint_as_float(u & 0x7FFFFFFFu) : __uint_as_float(~u);
}

__global__ void k_segmax(const float* __restrict__ score, const int* __restrict__ seg,
                         unsigned* __restrict__ smax, int Mn)
{
  const int i = blockIdx.x * blockDim.x + threadIdx.x;
  if (i < Mn) atomicMax(&smax[seg[i]], fenc(score[i]));
}

__global__ void k_segexp(const float* __restrict__ score, const int* __restrict__ seg,
                         const unsigned* __restrict__ smax, float* __restrict__ zs,
                         float* __restrict__ ssum, int Mn)
{
  const int i = blockIdx.x * blockDim.x + threadIdx.x;
  if (i >= Mn) return;
  const int n = seg[i];
  const float ev = expf(score[i] - fdec(smax[n]));
  zs[i] = ev;
  atomicAdd(&ssum[n], ev);
}

__global__ void k_segdiv(float* __restrict__ zs, const int* __restrict__ seg,
                         const float* __restrict__ ssum, int Mn)
{
  const int i = blockIdx.x * blockDim.x + threadIdx.x;
  if (i < Mn) zs[i] /= (ssum[seg[i]] + 1e-16f);
}

__global__ void k_pooled(const float* __restrict__ zs, const float* __restrict__ e,
                         const float* __restrict__ mean, const int* __restrict__ b_x,
                         float* __restrict__ pooled, int Mn)
{
  const int idx = blockIdx.x * blockDim.x + threadIdx.x;
  if (idx >= Mn * 400) return;
  const int i = idx / 400, c = idx % 400;
  const int n = b_x[i];
  const float v = (c < 200) ? e[(size_t)n * 200 + c] : mean[(size_t)n * 200 + (c - 200)];
  atomicAdd(&pooled[(size_t)n * 400 + c], zs[i] * v);
}

__global__ void k_colmean_sigmoid(const float* __restrict__ p, int rows, int Cw,
                                  float* __restrict__ out)
{
  __shared__ float red[256];
  const int c = blockIdx.x, t = threadIdx.x;
  float s = 0.0f;
  for (int r = t; r < rows; r += 256) s += p[(size_t)r * Cw + c];
  red[t] = s; __syncthreads();
  for (int st = 128; st > 0; st >>= 1) { if (t < st) red[t] += red[t + st]; __syncthreads(); }
  if (t == 0) out[c] = 1.0f / (1.0f + expf(-red[0] / (float)rows));
}

__global__ void k_matvec(const float* __restrict__ W, const float* __restrict__ x,
                         float* __restrict__ y, int cols)
{
  __shared__ float red[256];
  const int r = blockIdx.x, t = threadIdx.x;
  float s = 0.0f;
  for (int c = t; c < cols; c += 256) s += W[(size_t)r * cols + c] * x[c];
  red[t] = s; __syncthreads();
  for (int st = 128; st > 0; st >>= 1) { if (t < st) red[t] += red[t + st]; __syncthreads(); }
  if (t == 0) y[r] = red[0];
}

__global__ void k_rowdot2(const float* __restrict__ p1, const float* __restrict__ p2,
                          const float* __restrict__ tv, float* __restrict__ out,
                          int rows, int Cw)
{
  const int i = blockIdx.x * blockDim.x + threadIdx.x;
  if (i >= 2 * rows) return;
  const float* pr = (i < rows) ? (p1 + (size_t)i * Cw) : (p2 + (size_t)(i - rows) * Cw);
  float s = 0.0f;
  for (int c = 0; c < Cw; ++c) s += pr[c] * tv[c];
  out[i] = s;
}

__global__ void k_convkb(const float* __restrict__ ec, const float* __restrict__ rel,
                         const int* __restrict__ bi,
                         const float* __restrict__ cw, const float* __restrict__ cb,
                         const float* __restrict__ fw, const float* __restrict__ fb,
                         float* __restrict__ out)
{
  __shared__ float red[256];
  const int b = blockIdx.x, t = threadIdx.x;
  const int i0 = bi[b * 3 + 0], i1 = bi[b * 3 + 1], i2 = bi[b * 3 + 2];
  const float* h0 = ec  + (size_t)i0 * cD;
  const float* h1 = rel + (size_t)i1 * cD;
  const float* h2 = ec  + (size_t)i2 * cD;
  float acc = 0.0f;
  for (int idx = t; idx < cOC * cD; idx += 256) {
    const int oc = idx / cD, d = idx % cD;
    float h = h0[d] * cw[oc * 3 + 0] + h1[d] * cw[oc * 3 + 1] + h2[d] * cw[oc * 3 + 2]
            + cb[oc];
    h = fmaxf(h, 0.0f);
    acc += h * fw[idx];
  }
  red[t] = acc; __syncthreads();
  for (int st = 128; st > 0; st >>= 1) { if (t < st) red[t] += red[t + st]; __syncthreads(); }
  if (t == 0) out[b] = red[0] + fb[0];
}

// ---------------------------------------------------------------------------
// Host orchestration
// ---------------------------------------------------------------------------
static inline int ceil_div(int a, int b) { return (a + b - 1) / b; }

static void launch_gemm(const _Float16* Ah, const _Float16* BT,
                        float* C, int ldc, int M, int N, int Kp,
                        const float* bias, const float* addend, const int* add_perm,
                        int relu, hipStream_t s)
{
  dim3 grid(ceil_div(N, 16), ceil_div(M, 256));
  k_gemm_wmma<<<grid, 128, 0, s>>>(Ah, Kp, BT, C, ldc, M, N, Kp,
                                   bias, addend, add_perm, relu);
}

extern "C" void kernel_launch(void* const* d_in, const int* in_sizes, int n_in,
                              void* d_out, int out_size, void* d_ws, size_t ws_size,
                              hipStream_t stream)
{
  (void)in_sizes; (void)n_in; (void)out_size; (void)ws_size;

  const float* entity_emb = (const float*)d_in[0];
  const float* rel_emb    = (const float*)d_in[1];
  const int*   b_x   = (const int*)d_in[2];
  const int*   b_gi  = (const int*)d_in[3];
  const int*   esrc  = (const int*)d_in[4];
  const int*   edst  = (const int*)d_in[5];
  const float* ew    = (const float*)d_in[6];
  const int*   gsrc  = (const int*)d_in[7];
  const int*   gdst  = (const int*)d_in[8];
  const float* gw    = (const float*)d_in[9];
  const int*   perm  = (const int*)d_in[10];
  const int*   bi    = (const int*)d_in[11];
  const float* W1a = (const float*)d_in[12];
  const float* W1b = (const float*)d_in[13];
  const float* W2a = (const float*)d_in[14];
  const float* W2b = (const float*)d_in[15];
  const float* Wle = (const float*)d_in[16];
  const float* ble = (const float*)d_in[17];
  const float* Wlo = (const float*)d_in[18];
  const float* blo = (const float*)d_in[19];
  const float* Wg  = (const float*)d_in[20];
  const float* Wd  = (const float*)d_in[21];
  const float* cw  = (const float*)d_in[22];
  const float* cb  = (const float*)d_in[23];
  const float* fw  = (const float*)d_in[24];
  const float* fb  = (const float*)d_in[25];

  float* out_conv   = (float*)d_out;          // [B]
  float* out_local  = out_conv + cB;          // [2N]
  float* out_global = out_local + 2 * cN;     // [2N]

  constexpr int Kp400 = kpad(400), Kp256 = kpad(256), Kp200 = kpad(200);

  // workspace arena (~209M floats), 32B-aligned slices
  float* w = (float*)d_ws;
  size_t o = 0;
  auto take = [&](size_t n) { n = (n + 7) & ~(size_t)7; float* p = w + o; o += n; return p; };
  float* ENT  = take((size_t)cN * cD);
  float* RELn = take((size_t)cR * cD);
  float* EC   = take((size_t)cN * cD);
  float* ECS  = take((size_t)cN * cD);
  float* EG   = take((size_t)cN * cD);
  float* EGS  = take((size_t)cN * cD);
  float* MEAN = take((size_t)cN * cD);
  float* POOL = take((size_t)cN * 2 * cD);
  float* CNT  = take(cN);
  float* SCORE= take(cM);
  float* ZS   = take(cM);
  unsigned* SMAX = (unsigned*)take(cN);
  float* SSUM = take(cN);
  float* CV   = take(cHID);
  float* TV   = take(cHID);
  float* S1   = take((size_t)cM * 2 * cD);          // agg1 / agg2 / aggh / P1
  float* S2   = take((size_t)cM * cH1);             // h / hg / P2
  float* S4   = take((size_t)cM * cD);              // e
  _Float16* H16 = (_Float16*)take((size_t)cM * Kp400 / 2);   // f16 A staging
  _Float16* BT_W1a = (_Float16*)take((size_t)cH1  * Kp400 / 2);
  _Float16* BT_W1b = (_Float16*)take((size_t)cD   * Kp256 / 2);
  _Float16* BT_Wlo = (_Float16*)take((size_t)cD   * Kp400 / 2);
  _Float16* BT_W2a = (_Float16*)take((size_t)cH1  * Kp200 / 2);
  _Float16* BT_W2b = (_Float16*)take((size_t)cD   * Kp256 / 2);
  _Float16* BT_Wg  = (_Float16*)take((size_t)cHID * Kp200 / 2);

  // one-time weight prep (f32 row-major -> f16 K-padded column-major)
  auto prep = [&](const float* B, _Float16* BT, int K, int N) {
    const int Kp = kpad(K);
    k_prep_bT<<<ceil_div(N * Kp, 256), 256, 0, stream>>>(B, BT, K, N, Kp);
  };
  prep(W1a, BT_W1a, 2 * cD, cH1);
  prep(W1b, BT_W1b, cH1,    cD);
  prep(Wlo, BT_Wlo, 2 * cD, cD);
  prep(W2a, BT_W2a, cD,     cH1);
  prep(W2b, BT_W2b, cH1,    cD);
  prep(Wg,  BT_Wg,  cD,     cHID);

  auto cvtA = [&](const float* A, int rows, int K) {
    const int Kp = kpad(K);
    k_cvt_f16_pad<<<ceil_div(rows * Kp, 256), 256, 0, stream>>>(A, H16, rows, K, Kp);
  };

  auto mce = [&](const int* prm, float* outEC) {
    hipMemsetAsync(S1, 0, (size_t)cM * 2 * cD * sizeof(float), stream);
    k_agg_x<<<ceil_div(cE * 2 * cD, 256), 256, 0, stream>>>(
        ENT, prm, RELn, b_x, b_gi, esrc, edst, ew, S1, cE);
    // h = relu(agg1 @ W1a)  [M,400]@[400,256]
    cvtA(S1, cM, 2 * cD);
    launch_gemm(H16, BT_W1a, S2, cH1, cM, cH1, Kp400,
                nullptr, nullptr, nullptr, 1, stream);
    // agg2 (reuse S1)
    hipMemsetAsync(S1, 0, (size_t)cM * cH1 * sizeof(float), stream);
    k_agg<<<ceil_div(cE * cH1, 256), 256, 0, stream>>>(S2, cH1, esrc, edst, ew, S1, cE);
    // e = agg2 @ W1b  [M,256]@[256,200]
    cvtA(S1, cM, cH1);
    launch_gemm(H16, BT_W1b, S4, cD, cM, cD, Kp256,
                nullptr, nullptr, nullptr, 0, stream);
    // scatter-mean over entities
    hipMemsetAsync(CNT, 0, cN * sizeof(float), stream);
    hipMemsetAsync(MEAN, 0, (size_t)cN * cD * sizeof(float), stream);
    k_count<<<ceil_div(cM, 256), 256, 0, stream>>>(b_x, CNT, cM);
    k_scatter_add<<<ceil_div(cM * cD, 256), 256, 0, stream>>>(S4, b_x, MEAN, cM, cD);
    k_divcnt<<<ceil_div(cN * cD, 256), 256, 0, stream>>>(MEAN, CNT, cN, cD);
    // attention scores + segment softmax
    k_score<<<ceil_div(cM, 256), 256, 0, stream>>>(S4, MEAN, b_x, Wle, ble, SCORE, cM);
    hipMemsetAsync(SMAX, 0, cN * sizeof(unsigned), stream);
    hipMemsetAsync(SSUM, 0, cN * sizeof(float), stream);
    k_segmax<<<ceil_div(cM, 256), 256, 0, stream>>>(SCORE, b_x, SMAX, cM);
    k_segexp<<<ceil_div(cM, 256), 256, 0, stream>>>(SCORE, b_x, SMAX, ZS, SSUM, cM);
    k_segdiv<<<ceil_div(cM, 256), 256, 0, stream>>>(ZS, b_x, SSUM, cM);
    // attention-pooled
    hipMemsetAsync(POOL, 0, (size_t)cN * 2 * cD * sizeof(float), stream);
    k_pooled<<<ceil_div(cM * 2 * cD, 256), 256, 0, stream>>>(ZS, S4, MEAN, b_x, POOL, cM);
    // outEC = pooled @ Wlo + blo + ent[prm]  [N,400]@[400,200]
    cvtA(POOL, cN, 2 * cD);
    launch_gemm(H16, BT_Wlo, outEC, cD, cN, cD, Kp400,
                blo, ENT, prm, 0, stream);
  };

  auto gcn_big = [&](const float* hin, float* outE) {
    hipMemsetAsync(MEAN, 0, (size_t)cN * cD * sizeof(float), stream);
    k_agg<<<ceil_div(cNE * cD, 256), 256, 0, stream>>>(hin, cD, gsrc, gdst, gw, MEAN, cNE);
    cvtA(MEAN, cN, cD);
    launch_gemm(H16, BT_W2a, S2, cH1, cN, cH1, Kp200,
                nullptr, nullptr, nullptr, 1, stream);
    hipMemsetAsync(S1, 0, (size_t)cN * cH1 * sizeof(float), stream);
    k_agg<<<ceil_div(cNE * cH1, 256), 256, 0, stream>>>(S2, cH1, gsrc, gdst, gw, S1, cNE);
    cvtA(S1, cN, cH1);
    launch_gemm(H16, BT_W2b, outE, cD, cN, cD, Kp256,
                nullptr, nullptr, nullptr, 0, stream);
  };

  auto dgi = [&](const float* h1, const float* h2, float* outp) {
    float* P1 = S1;
    float* P2 = S2;
    cvtA(h1, cN, cD);
    launch_gemm(H16, BT_Wg, P1, cHID, cN, cHID, Kp200,
                nullptr, nullptr, nullptr, 1, stream);
    cvtA(h2, cN, cD);
    launch_gemm(H16, BT_Wg, P2, cHID, cN, cHID, Kp200,
                nullptr, nullptr, nullptr, 1, stream);
    k_colmean_sigmoid<<<cHID, 256, 0, stream>>>(P1, cN, cHID, CV);
    k_matvec<<<cHID, 256, 0, stream>>>(Wd, CV, TV, cHID);
    k_rowdot2<<<ceil_div(2 * cN, 256), 256, 0, stream>>>(P1, P2, TV, outp, cN, cHID);
  };

  // pipeline
  k_l2norm<<<cN, 256, 0, stream>>>(entity_emb, ENT, cD);
  k_l2norm<<<cR, 256, 0, stream>>>(rel_emb, RELn, cD);
  mce(nullptr, EC);
  mce(perm, ECS);
  gcn_big(EC, EG);
  gcn_big(ECS, EGS);
  dgi(EC, ECS, out_local);
  dgi(EG, EGS, out_global);
  k_convkb<<<cB, 256, 0, stream>>>(EC, RELn, bi, cw, cb, fw, fb, out_conv);
}
